// CrossAttention_37718402794152
// MI455X (gfx1250) — compile-verified, matrix operand order checked by reading, not run on hardware
//
#include <hip/hip_runtime.h>
#include <cstdint>

typedef __attribute__((ext_vector_type(16))) _Float16 v16h;
typedef __attribute__((ext_vector_type(8)))  _Float16 v8h;
typedef __attribute__((ext_vector_type(8)))  float    v8f;
typedef __attribute__((ext_vector_type(4)))  unsigned int u32x4;
typedef __attribute__((ext_vector_type(8)))  int          i32x8;
typedef __attribute__((ext_vector_type(4)))  int          i32x4;

#define NTOK 4800
#define CDIM 256
#define NH   8
#define HD_  32
#define DPAD 48
#define WPB  6                       // waves per block in attn_fund
#define SCALE 0.17677669529663687f   // 32^-0.5

__device__ __forceinline__ v16h cat8(v8h lo, v8h hi) {
  return __builtin_shufflevector(lo, hi, 0,1,2,3,4,5,6,7,8,9,10,11,12,13,14,15);
}

// Load a 16x32 f16 WMMA A/B fragment from a row-major matrix.
// Fragment rows row0..row0+15, K = koff..koff+31.
// Lane L holds row row0+(L&15); halves [0..7]=K(koff+8*(L>>4)..), [8..15]=K(+16).
__device__ __forceinline__ v16h ld_frag(const _Float16* base, int stride, int row0, int koff) {
  int l = threadIdx.x & 31;
  const _Float16* p = base + (size_t)(row0 + (l & 15)) * stride + koff + 8 * (l >> 4);
  v8h a = *(const v8h*)p;
  v8h b = *(const v8h*)(p + 16);
  return cat8(a, b);
}

__device__ __forceinline__ v8f wmma16(v16h a, v16h b, v8f c) {
  return __builtin_amdgcn_wmma_f32_16x16x32_f16(false, a, false, b, (short)0, c, false, false);
}

// Tensor Data Mover: stage a 2-D tile (8-byte elements) from global into LDS.
// Descriptor per CDNA5 ISA (groups 2/3 zero => 2-D tensor).
// This toolchain exposes the 6-arg builtin: (g0, g1, g2, g3, g4, cpol).
__device__ __forceinline__ void tdm_load_2d(unsigned lds_addr, const void* gaddr,
                                            unsigned tile_d0, unsigned tile_d1,
                                            unsigned long long stride0,
                                            unsigned tensor_d0, unsigned tensor_d1) {
  unsigned long long ga = (unsigned long long)(uintptr_t)gaddr;
  u32x4 g0;
  g0[0] = 1u;                                                  // count=1, user D#
  g0[1] = lds_addr;                                            // LDS byte address
  g0[2] = (unsigned)(ga & 0xffffffffu);                        // global_addr[31:0]
  g0[3] = (unsigned)((ga >> 32) & 0x01ffffffu) | (2u << 30);   // addr[56:32], type=2
  i32x8 g1;
  g1[0] = (int)(3u << 16);                                     // wg_mask=0, data_size=8B
  g1[1] = (int)((tensor_d0 & 0xffffu) << 16);                  // tensor_dim0[15:0]
  g1[2] = (int)(((tensor_d0 >> 16) & 0xffffu) | ((tensor_d1 & 0xffffu) << 16));
  g1[3] = (int)(((tensor_d1 >> 16) & 0xffffu) | (tile_d0 << 16)); // tile_dim0
  g1[4] = (int)(tile_d1 & 0xffffu);                            // tile_dim1, tile_dim2=0
  g1[5] = (int)(stride0 & 0xffffffffu);                        // dim0 stride[31:0]
  g1[6] = (int)((stride0 >> 32) & 0xffffu);                    // stride[47:32], s1=0
  g1[7] = 0;
  i32x4 z4 = {0, 0, 0, 0};
  i32x8 z8 = {0, 0, 0, 0, 0, 0, 0, 0};
  __builtin_amdgcn_tensor_load_to_lds(g0, g1, z4, z4, z8, 0);
}

// ---------------------------------------------------------------- f32 -> f16
__global__ void cvt_f16(const float* __restrict__ s, _Float16* __restrict__ d, int n) {
  int i = blockIdx.x * blockDim.x + threadIdx.x;
  if (i < n) d[i] = (_Float16)s[i];
}

// ------------------------------------------------------------ QKV projection
// Y = X @ W^T, scattered into per-head q/k/v [xi][h][n][d] f16.
__global__ __launch_bounds__(128) void qkv_gemm(const _Float16* __restrict__ X,
                                                const _Float16* __restrict__ W,
                                                _Float16* __restrict__ q,
                                                _Float16* __restrict__ k,
                                                _Float16* __restrict__ v) {
  int task = blockIdx.x * 4 + (threadIdx.x >> 5);   // [0, 28800)
  int xi = task / 14400; int rem = task % 14400;
  int ri = rem / 48, ci = rem % 48;
  int n0 = ri * 16, j0 = ci * 16;
  const _Float16* Xp = X + (size_t)xi * NTOK * CDIM;
  v8f acc = {};
  #pragma unroll
  for (int kc = 0; kc < CDIM; kc += 32) {
    v16h a = ld_frag(Xp, CDIM, n0, kc);
    v16h b = ld_frag(W, CDIM, j0, kc);
    acc = wmma16(a, b, acc);
  }
  int l = threadIdx.x & 31, lo = l & 15, hi = l >> 4;
  int j = j0 + lo;
  int t = j >> 8, hh = (j >> 5) & 7, d = j & 31;
  _Float16* dst = (t == 0 ? q : t == 1 ? k : v) + ((size_t)xi * NH + hh) * NTOK * HD_;
  #pragma unroll
  for (int r = 0; r < 8; ++r) {
    int n = n0 + r + 8 * hi;
    dst[(size_t)n * HD_ + d] = (_Float16)acc[r];
  }
}

// --------------------------------------------- row softmax stats of scale*QK^T
// (column stats = same kernel with Q and K swapped)
__global__ __launch_bounds__(128) void qk_stats(const _Float16* __restrict__ Qb,
                                                const _Float16* __restrict__ Kb,
                                                float* __restrict__ rmax,
                                                float* __restrict__ rsum) {
  int task = blockIdx.x * 4 + (threadIdx.x >> 5);   // [0, 1200)
  int h = task / 150, chunk = task % 150;
  int n0 = chunk * 32;
  const _Float16* Q = Qb + (size_t)h * NTOK * HD_;
  const _Float16* K = Kb + (size_t)h * NTOK * HD_;
  v16h aq0 = ld_frag(Q, HD_, n0, 0);
  v16h aq1 = ld_frag(Q, HD_, n0 + 16, 0);
  float mx[2][8], sm[2][8];
  #pragma unroll
  for (int i = 0; i < 2; ++i)
    #pragma unroll
    for (int r = 0; r < 8; ++r) { mx[i][r] = -__builtin_inff(); sm[i][r] = 0.f; }
  for (int m0 = 0; m0 < NTOK; m0 += 32) {
    v16h bk0 = ld_frag(K, HD_, m0, 0);
    v16h bk1 = ld_frag(K, HD_, m0 + 16, 0);
    v8f z = {};
    v8f s[2][2];
    s[0][0] = wmma16(aq0, bk0, z); s[0][1] = wmma16(aq0, bk1, z);
    s[1][0] = wmma16(aq1, bk0, z); s[1][1] = wmma16(aq1, bk1, z);
    #pragma unroll
    for (int i = 0; i < 2; ++i)
      #pragma unroll
      for (int r = 0; r < 8; ++r) {
        float x0 = SCALE * s[i][0][r], x1 = SCALE * s[i][1][r];
        float mn = fmaxf(mx[i][r], fmaxf(x0, x1));
        sm[i][r] = sm[i][r] * __expf(mx[i][r] - mn) + __expf(x0 - mn) + __expf(x1 - mn);
        mx[i][r] = mn;
      }
  }
  // merge (m,s) across the 16 lanes sharing the same rows
  #pragma unroll
  for (int off = 1; off < 16; off <<= 1)
    #pragma unroll
    for (int i = 0; i < 2; ++i)
      #pragma unroll
      for (int r = 0; r < 8; ++r) {
        float om = __shfl_xor(mx[i][r], off, 32);
        float os = __shfl_xor(sm[i][r], off, 32);
        float mn = fmaxf(mx[i][r], om);
        sm[i][r] = sm[i][r] * __expf(mx[i][r] - mn) + os * __expf(om - mn);
        mx[i][r] = mn;
      }
  int l = threadIdx.x & 31;
  if ((l & 15) == 0) {
    int hi = l >> 4;
    #pragma unroll
    for (int i = 0; i < 2; ++i)
      #pragma unroll
      for (int r = 0; r < 8; ++r) {
        int n = n0 + 16 * i + 8 * hi + r;
        rmax[(size_t)h * NTOK + n] = mx[i][r];
        rsum[(size_t)h * NTOK + n] = sm[i][r];
      }
  }
}

// ----------------------- build scaled, transposed concat(v,pos) f16 matrices
// vc1T[ph][c][n] = vc[n][c]/rowsum[n] ; vc2T[ph][c][n] = vc[n][c]/colsum[n]
__global__ void build_vct(const _Float16* __restrict__ vf,
                          const float* __restrict__ rs,
                          const float* __restrict__ cs,
                          _Float16* __restrict__ vc1T,
                          _Float16* __restrict__ vc2T) {
  int idx = blockIdx.x * blockDim.x + threadIdx.x;
  if (idx >= 2 * NH * NTOK * DPAD) return;
  int c = idx % DPAD;
  int n = (idx / DPAD) % NTOK;
  int ph = idx / (DPAD * NTOK);          // pi*8 + h ; v input index == pi
  float val;
  if (c < HD_) {
    val = (float)vf[((size_t)ph * NTOK + n) * HD_ + c];
  } else if (c < HD_ + 6) {
    int jr = n / 80, kc = n % 80;
    float y = -1.f + 2.f * (float)jr / 59.f;
    float x = -1.f + 2.f * (float)kc / 79.f;
    float p3 = y * (240.f / 517.f);
    float p4 = x * (360.f / 517.f);
    int cc = c - HD_;
    val = (cc == 0) ? p3 * p3 : (cc == 1) ? p4 * p4 : (cc == 2) ? p3 * p4 :
          (cc == 3) ? p3 : (cc == 4) ? p4 : 1.f;
  } else {
    val = 0.f;                            // pad 38 -> 48
  }
  size_t si = (size_t)ph * NTOK + n;
  size_t oi = ((size_t)ph * DPAD + c) * NTOK + n;
  vc1T[oi] = (_Float16)(val / rs[si]);
  vc2T[oi] = (_Float16)(val / cs[si]);
}

__global__ void zero_f32(float* __restrict__ p, int n) {
  int i = blockIdx.x * blockDim.x + threadIdx.x;
  if (i < n) p[i] = 0.f;
}

// ------------------------------- streaming F = Vc1~^T exp(2a-rm-cm) Vc2~
// 6 waves/block share one (problem,head); K and Vc2~ chunks are staged into
// LDS once per block by the Tensor Data Mover, double-buffered so the DMA of
// chunk i+1 overlaps WMMA work on chunk i (TENSORcnt + workgroup barriers).
__global__ __launch_bounds__(32 * WPB) void attn_fund(const _Float16* __restrict__ qf,
                                                      const _Float16* __restrict__ kf,
                                                      const _Float16* __restrict__ vc1T,
                                                      const _Float16* __restrict__ vc2T,
                                                      const float* __restrict__ rmax,
                                                      const float* __restrict__ cmax,
                                                      float* __restrict__ F) {
  __shared__ _Float16 Kst[2][32 * 32];        // 2 x 2KB
  __shared__ _Float16 V2st[2][48 * 32];       // 2 x 3KB
  __shared__ _Float16 plds[WPB][32 * 32];     // per-wave P transpose staging
  int wv = threadIdx.x >> 5;
  int ph = blockIdx.x / 25;                   // 16 (problem,head) pairs
  int chunk = (blockIdx.x % 25) * WPB + wv;   // [0,150)
  int pi = ph >> 3, h = ph & 7;
  int n0 = chunk * 32;
  const _Float16* Q   = qf + ((size_t)(1 - pi) * NH + h) * NTOK * HD_; // p0:q2 p1:q1
  const _Float16* K   = kf + ((size_t)pi * NH + h) * NTOK * HD_;       // p0:k1 p1:k2
  const _Float16* V1T = vc1T + (size_t)ph * DPAD * NTOK;
  const _Float16* V2T = vc2T + (size_t)ph * DPAD * NTOK;
  const float* rmp = rmax + (size_t)ph * NTOK;
  const float* cmp = cmax + (size_t)ph * NTOK;
  float* Fp = F + (size_t)ph * DPAD * DPAD;
  int l = threadIdx.x & 31, lo = l & 15, hi = l >> 4;
  bool leader = (wv == 0);

  v16h aq0 = ld_frag(Q, HD_, n0, 0);
  v16h aq1 = ld_frag(Q, HD_, n0 + 16, 0);
  float rm0[8], rm1[8];
  #pragma unroll
  for (int r = 0; r < 8; ++r) {
    rm0[r] = rmp[n0 + 8 * hi + r];
    rm1[r] = rmp[n0 + 16 + 8 * hi + r];
  }
  v8f t[2][3] = {};
  _Float16* my = plds[wv];
  const float sc2 = 2.f * SCALE;

  if (leader) {            // prologue: stage chunk 0 into buffer 0
    tdm_load_2d((unsigned)(uintptr_t)&Kst[0][0],  K,   256, 0, 256, 256, 1);
    tdm_load_2d((unsigned)(uintptr_t)&V2st[0][0], V2T, 8, 48, 1200, 1200, 48);
  }
  int buf = 0;
  for (int mi = 0; mi < 150; ++mi) {
    int m0 = mi * 32;
    __syncthreads();                                   // prior compute finished
    if (leader) __builtin_amdgcn_s_wait_tensorcnt(0);  // DMA for `buf` done
    __syncthreads();                                   // publish staged chunk
    if (leader && mi + 1 < 150) {                      // prefetch next chunk
      tdm_load_2d((unsigned)(uintptr_t)&Kst[buf ^ 1][0],
                  K + (size_t)(m0 + 32) * HD_, 256, 0, 256, 256, 1);
      tdm_load_2d((unsigned)(uintptr_t)&V2st[buf ^ 1][0],
                  V2T + (m0 + 32), 8, 48, 1200, 1200, 48);
    }
    const _Float16* Kc = Kst[buf];
    const _Float16* Vc = V2st[buf];
    v16h bk0 = ld_frag(Kc, 32, 0, 0);
    v16h bk1 = ld_frag(Kc, 32, 16, 0);
    v8f z = {};
    v8f s[2][2];
    s[0][0] = wmma16(aq0, bk0, z); s[0][1] = wmma16(aq0, bk1, z);
    s[1][0] = wmma16(aq1, bk0, z); s[1][1] = wmma16(aq1, bk1, z);
    float cm0 = cmp[m0 + lo], cm1 = cmp[m0 + 16 + lo];
    // P = exp(2a - rm - cm), staged f16 into LDS for lane transpose
    #pragma unroll
    for (int i = 0; i < 2; ++i) {
      const float* rmv = i ? rm1 : rm0;
      #pragma unroll
      for (int j = 0; j < 2; ++j) {
        float cmv = j ? cm1 : cm0;
        #pragma unroll
        for (int r = 0; r < 8; ++r) {
          float pv = __expf(sc2 * s[i][j][r] - rmv[r] - cmv);
          my[(16 * i + r + 8 * hi) * 32 + 16 * j + lo] = (_Float16)pv;
        }
      }
    }
    // reload P as A fragments (same wave, DS ops are in order)
    const _Float16* p0 = my + (size_t)lo * 32 + 8 * hi;
    v16h ap0 = cat8(*(const v8h*)p0, *(const v8h*)(p0 + 16));
    const _Float16* p1 = my + (size_t)(16 + lo) * 32 + 8 * hi;
    v16h ap1 = cat8(*(const v8h*)p1, *(const v8h*)(p1 + 16));

    v16h bv0 = ld_frag(Vc, 32, 0, 0);
    v16h bv1 = ld_frag(Vc, 32, 16, 0);
    v16h bv2 = ld_frag(Vc, 32, 32, 0);
    t[0][0] = wmma16(ap0, bv0, t[0][0]);
    t[0][1] = wmma16(ap0, bv1, t[0][1]);
    t[0][2] = wmma16(ap0, bv2, t[0][2]);
    t[1][0] = wmma16(ap1, bv0, t[1][0]);
    t[1][1] = wmma16(ap1, bv1, t[1][1]);
    t[1][2] = wmma16(ap1, bv2, t[1][2]);
    buf ^= 1;
  }

  // Repack T (C/D layout) into B fragments lane-locally: K=n_local, N=c.
  v16h bt[3];
  #pragma unroll
  for (int cj = 0; cj < 3; ++cj) {
    v16h b;
    #pragma unroll
    for (int r = 0; r < 8; ++r) {
      b[r]     = (_Float16)t[0][cj][r];
      b[8 + r] = (_Float16)t[1][cj][r];
    }
    bt[cj] = b;
  }
  v8f z = {};
  #pragma unroll
  for (int ci = 0; ci < 3; ++ci) {
    v16h av = ld_frag(V1T, NTOK, 16 * ci, n0);
    #pragma unroll
    for (int cj = 0; cj < 3; ++cj) {
      v8f f = wmma16(av, bt[cj], z);
      #pragma unroll
      for (int r = 0; r < 8; ++r)
        unsafeAtomicAdd(&Fp[(size_t)(16 * ci + r + 8 * hi) * DPAD + 16 * cj + lo], f[r]);
    }
  }
}

// ------------------------------------------------- out = f @ proj_w^T + b
__global__ void proj_out(const float* __restrict__ F,
                         const float* __restrict__ pw,
                         const float* __restrict__ pb,
                         float* __restrict__ out) {
  int idx = blockIdx.x * blockDim.x + threadIdx.x;
  if (idx >= 2 * 38 * CDIM) return;
  int o = idx % CDIM;
  int d = (idx / CDIM) % 38;
  int po = idx / (CDIM * 38);
  int pi = (po == 0) ? 1 : 0;            // reference returns (fundamental_2, fundamental_1)
  const float* Fp = F + (size_t)pi * NH * DPAD * DPAD;
  float acc = pb[o];
  for (int h = 0; h < NH; ++h)
    #pragma unroll
    for (int c = 0; c < 38; ++c)
      acc += Fp[(size_t)h * DPAD * DPAD + c * DPAD + d] * pw[(size_t)o * 304 + h * 38 + c];
  out[idx] = acc;
}

extern "C" void kernel_launch(void* const* d_in, const int* in_sizes, int n_in,
                              void* d_out, int out_size, void* d_ws, size_t ws_size,
                              hipStream_t stream) {
  (void)in_sizes; (void)n_in; (void)out_size; (void)ws_size;
  const float* x1     = (const float*)d_in[0];
  const float* x2     = (const float*)d_in[1];
  const float* qkv_w  = (const float*)d_in[2];
  const float* proj_w = (const float*)d_in[3];
  const float* proj_b = (const float*)d_in[4];
  float* out = (float*)d_out;

  char* ws = (char*)d_ws;
  size_t off = 0;
  auto take = [&](size_t bytes) -> char* {
    char* p = ws + off; off += (bytes + 255) & ~(size_t)255; return p;
  };
  _Float16* Xf16 = (_Float16*)take((size_t)2 * NTOK * CDIM * 2);
  _Float16* Wf16 = (_Float16*)take((size_t)3 * CDIM * CDIM * 2);
  _Float16* qf   = (_Float16*)take((size_t)2 * NH * NTOK * HD_ * 2);
  _Float16* kf   = (_Float16*)take((size_t)2 * NH * NTOK * HD_ * 2);
  _Float16* vf   = (_Float16*)take((size_t)2 * NH * NTOK * HD_ * 2);
  float* rmax = (float*)take((size_t)2 * NH * NTOK * 4);
  float* rsum = (float*)take((size_t)2 * NH * NTOK * 4);
  float* cmax = (float*)take((size_t)2 * NH * NTOK * 4);
  float* csum = (float*)take((size_t)2 * NH * NTOK * 4);
  _Float16* vc1T = (_Float16*)take((size_t)2 * NH * DPAD * NTOK * 2);
  _Float16* vc2T = (_Float16*)take((size_t)2 * NH * DPAD * NTOK * 2);
  float* Fws = (float*)take((size_t)2 * NH * DPAD * DPAD * 4);

  cvt_f16<<<(NTOK * CDIM + 255) / 256, 256, 0, stream>>>(x1, Xf16, NTOK * CDIM);
  cvt_f16<<<(NTOK * CDIM + 255) / 256, 256, 0, stream>>>(x2, Xf16 + (size_t)NTOK * CDIM, NTOK * CDIM);
  cvt_f16<<<(3 * CDIM * CDIM + 255) / 256, 256, 0, stream>>>(qkv_w, Wf16, 3 * CDIM * CDIM);

  qkv_gemm<<<7200, 128, 0, stream>>>(Xf16, Wf16, qf, kf, vf);

  const size_t HOFF = (size_t)NH * NTOK * HD_;   // per-input offset in q/k/v
  const size_t SOFF = (size_t)NH * NTOK;         // per-problem offset in stats
  // pi=0: fundamental_1 = (q2,k1,v1) ; pi=1: fundamental_2 = (q1,k2,v2)
  qk_stats<<<300, 128, 0, stream>>>(qf + HOFF, kf,        rmax,        rsum);        // p0 rows
  qk_stats<<<300, 128, 0, stream>>>(kf,        qf + HOFF, cmax,        csum);        // p0 cols
  qk_stats<<<300, 128, 0, stream>>>(qf,        kf + HOFF, rmax + SOFF, rsum + SOFF); // p1 rows
  qk_stats<<<300, 128, 0, stream>>>(kf + HOFF, qf,        cmax + SOFF, csum + SOFF); // p1 cols

  build_vct<<<(2 * NH * NTOK * DPAD + 255) / 256, 256, 0, stream>>>(vf, rsum, csum, vc1T, vc2T);
  zero_f32<<<(2 * NH * DPAD * DPAD + 255) / 256, 256, 0, stream>>>(Fws, 2 * NH * DPAD * DPAD);

  attn_fund<<<400, 32 * WPB, 0, stream>>>(qf, kf, vc1T, vc2T, rmax, cmax, Fws);

  proj_out<<<(2 * 38 * CDIM + 255) / 256, 256, 0, stream>>>(Fws, proj_w, proj_b, out);
}
